// GraspNet_3702261809768
// MI455X (gfx1250) — compile-verified, matrix operand order checked by reading, not run on hardware
//
#include <hip/hip_runtime.h>
#include <hip/hip_bf16.h>

typedef __attribute__((ext_vector_type(16))) _Float16 v16h;
typedef __attribute__((ext_vector_type(8)))  float    v8f;

#define BB 8
#define NN 4096
#define N1C 819
#define N2C 204
#define KNBR 64

// Async global->LDS staging (gfx1250 ASYNCcnt path). ldsOff = LDS byte address
// (low 32 bits of the generic shared-space pointer), gptr = 64-bit global addr.
__device__ __forceinline__ void async_g2l_b32(unsigned int ldsOff, const float* gptr)
{
    asm volatile("global_load_async_to_lds_b32 %0, %1, off"
                 :: "v"(ldsOff), "v"((unsigned long long)(uintptr_t)gptr)
                 : "memory");
}
__device__ __forceinline__ void wait_asynccnt0()
{
    asm volatile("s_wait_asynccnt 0" ::: "memory");
}

// ---------------------------------------------------------------------------
// Generic GEMM: C[M,N] = act(A[M,K] * W[K,N] + bias), f16 WMMA, f32 accum.
// Block = 256 threads (8 waves). Block tile 64x64; wave tile 16x32 (2 wmma).
// Interior tiles staged via global_load_async_to_lds_b32.
// ---------------------------------------------------------------------------
__global__ void gemm_bias_act(const float* __restrict__ A, const float* __restrict__ W,
                              const float* __restrict__ bias, float* __restrict__ C,
                              int M, int K, int N, int relu)
{
    __shared__ float Asf[64 * 33];          // [row][k], stride 33
    __shared__ float Bsf[64 * 33];          // transposed: [n][k], stride 33

    const int tid  = threadIdx.x;
    const int lane = tid & 31;
    const int w    = tid >> 5;
    const int wr   = w & 3;                 // wave row-tile 0..3
    const int wc   = w >> 2;                // wave col-group 0..1 (32 cols each)
    const int l16  = lane & 15;
    const int kg   = lane >> 4;
    const int rowBase = blockIdx.x * 64;
    const int colBase = blockIdx.y * 64;

    v8f acc0 = {}, acc1 = {};

    for (int k0 = 0; k0 < K; k0 += 32) {
        const bool fullK = (k0 + 32 <= K);
        const bool intA  = fullK && (rowBase + 64 <= M);
        const bool intB  = fullK && (colBase + 64 <= N);

        if (intA) {
            #pragma unroll
            for (int i = 0; i < 8; ++i) {
                int e = tid + i * 256;              // 64x32 = 2048 elements
                int r = e >> 5, kk = e & 31;
                async_g2l_b32((unsigned int)(uintptr_t)&Asf[r * 33 + kk],
                              &A[(size_t)(rowBase + r) * K + k0 + kk]);
            }
        } else {
            #pragma unroll
            for (int i = 0; i < 8; ++i) {
                int e = tid + i * 256;
                int r = e >> 5, kk = e & 31;
                int gr = rowBase + r, gk = k0 + kk;
                Asf[r * 33 + kk] = (gr < M && gk < K) ? A[(size_t)gr * K + gk] : 0.f;
            }
        }
        if (intB) {
            #pragma unroll
            for (int i = 0; i < 8; ++i) {
                int e = tid + i * 256;              // 32k x 64n = 2048 elements
                int kk = e >> 6, n = e & 63;
                async_g2l_b32((unsigned int)(uintptr_t)&Bsf[n * 33 + kk],
                              &W[(size_t)(k0 + kk) * N + colBase + n]);
            }
        } else {
            #pragma unroll
            for (int i = 0; i < 8; ++i) {
                int e = tid + i * 256;
                int kk = e >> 6, n = e & 63;
                int gk = k0 + kk, gn = colBase + n;
                Bsf[n * 33 + kk] = (gk < K && gn < N) ? W[(size_t)gk * N + gn] : 0.f;
            }
        }
        wait_asynccnt0();
        __syncthreads();

        union { v16h v; _Float16 h[16]; } a, b0, b1;
        const float* ar  = &Asf[(wr * 16 + l16) * 33];
        const float* br0 = &Bsf[(wc * 32 + l16) * 33];
        const float* br1 = &Bsf[(wc * 32 + 16 + l16) * 33];
        #pragma unroll
        for (int i = 0; i < 8; ++i) {
            a.h[i]      = (_Float16)ar[kg * 8 + i];
            a.h[8 + i]  = (_Float16)ar[16 + kg * 8 + i];
            b0.h[i]     = (_Float16)br0[kg * 8 + i];
            b0.h[8 + i] = (_Float16)br0[16 + kg * 8 + i];
            b1.h[i]     = (_Float16)br1[kg * 8 + i];
            b1.h[8 + i] = (_Float16)br1[16 + kg * 8 + i];
        }
        acc0 = __builtin_amdgcn_wmma_f32_16x16x32_f16(false, a.v, false, b0.v,
                                                      (short)0, acc0, false, false);
        acc1 = __builtin_amdgcn_wmma_f32_16x16x32_f16(false, a.v, false, b1.v,
                                                      (short)0, acc1, false, false);
        __syncthreads();
    }

    const int col0 = colBase + wc * 32 + l16;
    const int col1 = col0 + 16;
    #pragma unroll
    for (int r = 0; r < 8; ++r) {
        const int row = rowBase + wr * 16 + r + 8 * kg;
        if (row < M) {
            if (col0 < N) {
                float v = acc0[r] + (bias ? bias[col0] : 0.f);
                if (relu) v = v > 0.f ? v : 0.f;
                C[(size_t)row * N + col0] = v;
            }
            if (col1 < N) {
                float v = acc1[r] + (bias ? bias[col1] : 0.f);
                if (relu) v = v > 0.f ? v : 0.f;
                C[(size_t)row * N + col1] = v;
            }
        }
    }
}

// ---------------------------------------------------------------------------
// Fused SA module: gather neighbor features -> 3-layer MLP (WMMA, act in LDS)
// -> masked max-pool over 65 neighbors. One block (256 thr) per centroid.
// K-loop split into unguarded full tiles + one guarded tail tile.
// ---------------------------------------------------------------------------
#define SA_ROWS   80      // 65 neighbor rows padded to 5 row-tiles of 16
#define SA_STRIDE 264     // halves per activation row (max Cout 256 + pad)

__global__ void sa_mlp_maxpool(const float* x, int Cx,
                               const float* pos,
                               const float* posc,
                               const int*   __restrict__ nbr,
                               const float* __restrict__ d2s,
                               float r2, int Nsrc, int m,
                               const float* __restrict__ W0, const float* __restrict__ B0, int C1,
                               const float* __restrict__ W1, const float* __restrict__ B1, int C2,
                               const float* __restrict__ W2, const float* __restrict__ B2, int C3,
                               float* __restrict__ out)
{
    extern __shared__ _Float16 smem[];
    _Float16* bufA = smem;
    _Float16* bufB = smem + SA_ROWS * SA_STRIDE;

    const int ci  = blockIdx.x % m;
    const int b   = blockIdx.x / m;
    const int tid = threadIdx.x;
    const int lane = tid & 31;
    const int wv   = tid >> 5;
    const int l16  = lane & 15;
    const int kg   = lane >> 4;
    const int Cin  = Cx + 3;

    const float* posB = pos + (size_t)b * Nsrc * 3;
    const float* xB   = x   + (size_t)b * Nsrc * Cx;
    const float* pcp  = posc + ((size_t)b * m + ci) * 3;
    const int*   nb   = nbr  + ((size_t)b * m + ci) * KNBR;
    const float  pcx = pcp[0], pcy = pcp[1], pcz = pcp[2];

    for (int i = tid; i < SA_ROWS * SA_STRIDE; i += blockDim.x) bufA[i] = (_Float16)0.f;
    __syncthreads();

    for (int r = tid; r < 65; r += blockDim.x) {
        int src = (r == 64) ? ci : nb[r];            // reference appends arange(m)
        _Float16* dst = bufA + r * SA_STRIDE;
        for (int c = 0; c < Cx; ++c) dst[c] = (_Float16)xB[(size_t)src * Cx + c];
        dst[Cx + 0] = (_Float16)(posB[src * 3 + 0] - pcx);
        dst[Cx + 1] = (_Float16)(posB[src * 3 + 1] - pcy);
        dst[Cx + 2] = (_Float16)(posB[src * 3 + 2] - pcz);
    }
    __syncthreads();

    _Float16* in = bufA;
    _Float16* ot = bufB;
    int cin = Cin;
    for (int layer = 0; layer < 3; ++layer) {
        const float* Wl = (layer == 0) ? W0 : (layer == 1) ? W1 : W2;
        const float* Bl = (layer == 0) ? B0 : (layer == 1) ? B1 : B2;
        const int cout  = (layer == 0) ? C1 : (layer == 1) ? C2 : C3;
        const int relu  = layer < 2;
        const int ntiles = 5 * (cout >> 4);
        const int kfull  = cin & ~31;                // unguarded part

        for (int t = wv; t < ntiles; t += 8) {
            const int rt = t % 5, ct = t / 5;
            const int n  = ct * 16 + l16;
            v8f acc = {};
            for (int k0 = 0; k0 < kfull; k0 += 32) {
                union { v16h v; _Float16 h[16]; } a, bf;
                const _Float16* ar = in + (rt * 16 + l16) * SA_STRIDE + k0;
                const float*    wp = Wl + (size_t)k0 * cout + n;
                #pragma unroll
                for (int i = 0; i < 8; ++i) {
                    a.h[i]      = ar[kg * 8 + i];
                    a.h[8 + i]  = ar[16 + kg * 8 + i];
                    bf.h[i]     = (_Float16)wp[(size_t)(kg * 8 + i) * cout];
                    bf.h[8 + i] = (_Float16)wp[(size_t)(16 + kg * 8 + i) * cout];
                }
                acc = __builtin_amdgcn_wmma_f32_16x16x32_f16(false, a.v, false, bf.v,
                                                             (short)0, acc, false, false);
            }
            if (kfull < cin) {                       // guarded tail tile
                union { v16h v; _Float16 h[16]; } a, bf;
                const _Float16* ar = in + (rt * 16 + l16) * SA_STRIDE + kfull;
                #pragma unroll
                for (int i = 0; i < 16; ++i) {
                    int kk = kfull + ((i < 8) ? (kg * 8 + i) : (16 + kg * 8 + (i - 8)));
                    a.h[i]  = ar[(i < 8) ? (kg * 8 + i) : (16 + kg * 8 + (i - 8))];
                    bf.h[i] = (kk < cin) ? (_Float16)Wl[(size_t)kk * cout + n] : (_Float16)0.f;
                }
                acc = __builtin_amdgcn_wmma_f32_16x16x32_f16(false, a.v, false, bf.v,
                                                             (short)0, acc, false, false);
            }
            const float bv = Bl[n];
            #pragma unroll
            for (int r = 0; r < 8; ++r) {
                int row = rt * 16 + r + 8 * kg;
                float v = acc[r] + bv;
                if (relu) v = v > 0.f ? v : 0.f;
                ot[row * SA_STRIDE + n] = (_Float16)v;
            }
        }
        __syncthreads();
        _Float16* tmp = in; in = ot; ot = tmp;
        cin = cout;
    }

    const float* dd = d2s + ((size_t)b * m + ci) * KNBR;
    for (int c = tid; c < cin; c += blockDim.x) {
        float best = -1e30f;
        for (int r = 0; r < 65; ++r) {
            bool valid = (r == 64) || (dd[r] <= r2);
            if (valid) best = fmaxf(best, (float)in[r * SA_STRIDE + c]);
        }
        out[((size_t)b * m + ci) * cin + c] = best;
    }
}

// ---------------------------------------------------------------------------
// Farthest point sampling. One block per sample; dmin lives in LDS.
// ---------------------------------------------------------------------------
__global__ void fps_kernel(const float* __restrict__ pos, int N, int m,
                           float* __restrict__ posc, int* __restrict__ idxOut)
{
    __shared__ float dmin[4096];
    __shared__ float rv[256];
    __shared__ int   ri[256];
    __shared__ int   sLast;
    const int b = blockIdx.x, tid = threadIdx.x;
    const float* P = pos + (size_t)b * N * 3;

    for (int i = tid; i < N; i += 256) dmin[i] = 3.4e38f;
    if (tid == 0) sLast = 0;
    __syncthreads();

    for (int t = 0; t < m; ++t) {
        const int last = sLast;
        if (tid == 0) idxOut[(size_t)b * m + t] = last;   // scan records pre-update last
        const float lx = P[last * 3], ly = P[last * 3 + 1], lz = P[last * 3 + 2];
        float bv = -1.f; int bi = 0;
        for (int i = tid; i < N; i += 256) {
            float dx = P[i * 3] - lx, dy = P[i * 3 + 1] - ly, dz = P[i * 3 + 2] - lz;
            float dm = fminf(dmin[i], dx * dx + dy * dy + dz * dz);
            dmin[i] = dm;
            if (dm > bv) { bv = dm; bi = i; }
        }
        rv[tid] = bv; ri[tid] = bi;
        __syncthreads();
        for (int s = 128; s > 0; s >>= 1) {
            if (tid < s && (rv[tid + s] > rv[tid] ||
                            (rv[tid + s] == rv[tid] && ri[tid + s] < ri[tid]))) {
                rv[tid] = rv[tid + s]; ri[tid] = ri[tid + s];
            }
            __syncthreads();
        }
        if (tid == 0) sLast = ri[0];
        __syncthreads();
    }
    for (int t = tid; t < m; t += 256) {
        int id = idxOut[(size_t)b * m + t];
        posc[((size_t)b * m + t) * 3 + 0] = P[id * 3 + 0];
        posc[((size_t)b * m + t) * 3 + 1] = P[id * 3 + 1];
        posc[((size_t)b * m + t) * 3 + 2] = P[id * 3 + 2];
    }
}

// ---------------------------------------------------------------------------
// K=64 nearest neighbors by iterative min-extraction. One block per centroid.
// ---------------------------------------------------------------------------
__global__ void topk_kernel(const float* __restrict__ pos, const float* __restrict__ posc,
                            int Nsrc, int m,
                            float* __restrict__ d2out, int* __restrict__ nbrOut)
{
    __shared__ float d2[4096];
    __shared__ float rv[256];
    __shared__ int   ri[256];
    const int ci = blockIdx.x % m;
    const int b  = blockIdx.x / m;
    const int tid = threadIdx.x;
    const float* P  = pos + (size_t)b * Nsrc * 3;
    const float* pc = posc + ((size_t)b * m + ci) * 3;
    const float cx = pc[0], cy = pc[1], cz = pc[2];

    for (int i = tid; i < Nsrc; i += 256) {
        float dx = P[i * 3] - cx, dy = P[i * 3 + 1] - cy, dz = P[i * 3 + 2] - cz;
        d2[i] = dx * dx + dy * dy + dz * dz;
    }
    __syncthreads();

    for (int k = 0; k < KNBR; ++k) {
        float bv = 3.4e38f; int bi = 0;
        for (int i = tid; i < Nsrc; i += 256)
            if (d2[i] < bv) { bv = d2[i]; bi = i; }
        rv[tid] = bv; ri[tid] = bi;
        __syncthreads();
        for (int s = 128; s > 0; s >>= 1) {
            if (tid < s && (rv[tid + s] < rv[tid] ||
                            (rv[tid + s] == rv[tid] && ri[tid + s] < ri[tid]))) {
                rv[tid] = rv[tid + s]; ri[tid] = ri[tid + s];
            }
            __syncthreads();
        }
        if (tid == 0) {
            d2out[((size_t)b * m + ci) * KNBR + k]  = rv[0];
            nbrOut[((size_t)b * m + ci) * KNBR + k] = ri[0];
            d2[ri[0]] = 3.4e38f;
        }
        __syncthreads();
    }
}

// ---------------------------------------------------------------------------
// Small utility kernels
// ---------------------------------------------------------------------------
__global__ void concat2_kernel(const float* __restrict__ A, int Ca,
                               const float* __restrict__ Bv, int Cb,
                               float* __restrict__ dst, int rows)
{
    const int idx = blockIdx.x * 256 + threadIdx.x;
    const int C = Ca + Cb;
    if (idx >= rows * C) return;
    const int r = idx / C, c = idx % C;
    dst[idx] = (c < Ca) ? A[(size_t)r * Ca + c] : Bv[(size_t)r * Cb + (c - Ca)];
}

__global__ void concat_bcast_kernel(const float* __restrict__ perRow, int Ca,
                                    const float* __restrict__ bc, int Cb,
                                    int rowsPerSample, int bcFirst,
                                    float* __restrict__ dst, int totalRows)
{
    const int idx = blockIdx.x * 256 + threadIdx.x;
    const int C = Ca + Cb;
    if (idx >= totalRows * C) return;
    const int r = idx / C, c = idx % C;
    const int b = r / rowsPerSample;
    float v;
    if (bcFirst) v = (c < Cb) ? bc[(size_t)b * Cb + c] : perRow[(size_t)r * Ca + (c - Cb)];
    else         v = (c < Ca) ? perRow[(size_t)r * Ca + c] : bc[(size_t)b * Cb + (c - Ca)];
    dst[idx] = v;
}

__global__ void rowmax_kernel(const float* __restrict__ X, int rowsPerSample, int C,
                              float* __restrict__ out)
{
    const int b = blockIdx.x;
    for (int c = threadIdx.x; c < C; c += blockDim.x) {
        float best = -3.4e38f;
        for (int r = 0; r < rowsPerSample; ++r)
            best = fmaxf(best, X[((size_t)b * rowsPerSample + r) * C + c]);
        out[(size_t)b * C + c] = best;
    }
}

__global__ void knn3_interp_kernel(const float* __restrict__ feat,
                                   const float* __restrict__ psrc,
                                   const float* __restrict__ pdst,
                                   int S, int D, int C, int Btot,
                                   float* __restrict__ out)
{
    const int idx = blockIdx.x * 256 + threadIdx.x;
    if (idx >= Btot * D) return;
    const int b = idx / D, d = idx % D;
    const float* ps = psrc + (size_t)b * S * 3;
    const float* pd = pdst + ((size_t)b * D + d) * 3;
    const float px = pd[0], py = pd[1], pz = pd[2];
    float bd0 = 3.4e38f, bd1 = 3.4e38f, bd2 = 3.4e38f;
    int   bi0 = 0, bi1 = 0, bi2 = 0;
    for (int s = 0; s < S; ++s) {
        float dx = ps[s * 3] - px, dy = ps[s * 3 + 1] - py, dz = ps[s * 3 + 2] - pz;
        float dv = dx * dx + dy * dy + dz * dz;
        if (dv < bd0)      { bd2 = bd1; bi2 = bi1; bd1 = bd0; bi1 = bi0; bd0 = dv; bi0 = s; }
        else if (dv < bd1) { bd2 = bd1; bi2 = bi1; bd1 = dv; bi1 = s; }
        else if (dv < bd2) { bd2 = dv; bi2 = s; }
    }
    const float w0 = 1.f / fmaxf(bd0, 1e-16f);
    const float w1 = 1.f / fmaxf(bd1, 1e-16f);
    const float w2 = 1.f / fmaxf(bd2, 1e-16f);
    const float wsum = w0 + w1 + w2;
    const float* F = feat + (size_t)b * S * C;
    float* O = out + ((size_t)b * D + d) * C;
    for (int c = 0; c < C; ++c)
        O[c] = (F[(size_t)bi0 * C + c] * w0 + F[(size_t)bi1 * C + c] * w1 +
                F[(size_t)bi2 * C + c] * w2) / wsum;
}

__global__ void gather_ap_kernel(const float* __restrict__ pos, const int* __restrict__ apidx,
                                 int N, float* __restrict__ ap, int Bn)
{
    const int b = blockIdx.x * blockDim.x + threadIdx.x;
    if (b >= Bn) return;
    const int i = apidx[b];
    ap[b * 3 + 0] = pos[((size_t)b * N + i) * 3 + 0];
    ap[b * 3 + 1] = pos[((size_t)b * N + i) * 3 + 1];
    ap[b * 3 + 2] = pos[((size_t)b * N + i) * 3 + 2];
}

__global__ void top2_cond_kernel(const float* __restrict__ h, const float* __restrict__ pos,
                                 const float* __restrict__ ap, int N,
                                 float* __restrict__ cond, float* __restrict__ gp)
{
    __shared__ float rv[256];
    __shared__ int   ri[256];
    __shared__ int   win;
    const int b = blockIdx.x, tid = threadIdx.x;
    const float* H = h + (size_t)b * N;
    int gi0 = -1, gi1 = -1;
    for (int pass = 0; pass < 2; ++pass) {
        float bv = -3.4e38f; int bi = 0;
        for (int i = tid; i < N; i += 256) {
            if (pass == 1 && i == gi0) continue;
            float v = H[i];
            if (v > bv) { bv = v; bi = i; }
        }
        rv[tid] = bv; ri[tid] = bi;
        __syncthreads();
        for (int s = 128; s > 0; s >>= 1) {
            if (tid < s && (rv[tid + s] > rv[tid] ||
                            (rv[tid + s] == rv[tid] && ri[tid + s] < ri[tid]))) {
                rv[tid] = rv[tid + s]; ri[tid] = ri[tid + s];
            }
            __syncthreads();
        }
        if (tid == 0) win = ri[0];
        __syncthreads();
        if (pass == 0) gi0 = win; else gi1 = win;
        __syncthreads();
    }
    if (tid == 0) {
        const float* P = pos + (size_t)b * N * 3;
        for (int j = 0; j < 3; ++j) {
            gp[b * 6 + j]     = P[gi0 * 3 + j];
            gp[b * 6 + 3 + j] = P[gi1 * 3 + j];
            cond[b * 9 + j]     = ap[b * 3 + j];
            cond[b * 9 + 3 + j] = P[gi0 * 3 + j];
            cond[b * 9 + 6 + j] = P[gi1 * 3 + j];
        }
    }
}

__global__ void grasp_finalize_kernel(const float* __restrict__ trans,
                                      const float* __restrict__ gp,
                                      float* __restrict__ out, int Bn)
{
    const int b = blockIdx.x * blockDim.x + threadIdx.x;
    if (b >= Bn) return;
    float p1[3], p2[3], t[3];
    for (int j = 0; j < 3; ++j) { p1[j] = gp[b * 6 + j]; p2[j] = gp[b * 6 + 3 + j]; t[j] = trans[b * 3 + j]; }
    float z[3], mid[3], xa[3], y[3];
    float zn = 0.f;
    for (int j = 0; j < 3; ++j) { z[j] = p2[j] - p1[j]; zn += z[j] * z[j]; mid[j] = 0.5f * (p1[j] + p2[j]); }
    zn = sqrtf(zn);
    for (int j = 0; j < 3; ++j) z[j] /= zn;
    float xn = 0.f;
    for (int j = 0; j < 3; ++j) { xa[j] = t[j] - mid[j]; xn += xa[j] * xa[j]; }
    const float glen = sqrtf(xn);
    for (int j = 0; j < 3; ++j) xa[j] /= glen;
    y[0] = z[1] * xa[2] - z[2] * xa[1];
    y[1] = z[2] * xa[0] - z[0] * xa[2];
    y[2] = z[0] * xa[1] - z[1] * xa[0];
    float yn = sqrtf(y[0] * y[0] + y[1] * y[1] + y[2] * y[2]);
    for (int j = 0; j < 3; ++j) y[j] /= yn;
    float G[16] = {0};
    for (int i = 0; i < 3; ++i) {
        G[i * 4 + 0] = xa[i]; G[i * 4 + 1] = y[i]; G[i * 4 + 2] = z[i]; G[i * 4 + 3] = t[i];
    }
    G[15] = 1.f;
    for (int j = 0; j < 16; ++j) out[b * 16 + j] = G[j];
    out[Bn * 16 + b]      = glen;
    out[Bn * 16 + Bn + b] = z[0] * xa[0] + z[1] * xa[1] + z[2] * xa[2];
}

// ---------------------------------------------------------------------------
// Host launch
// ---------------------------------------------------------------------------
extern "C" void kernel_launch(void* const* d_in, const int* in_sizes, int n_in,
                              void* d_out, int out_size, void* d_ws, size_t ws_size,
                              hipStream_t stream)
{
    (void)in_sizes; (void)n_in; (void)out_size; (void)ws_size;

    const float* POS   = (const float*)d_in[0];
    const int*   APIDX = (const int*)d_in[1];
    int p = 2;
    const float *sa1W0=(const float*)d_in[p++], *sa1b0=(const float*)d_in[p++];
    const float *sa1W1=(const float*)d_in[p++], *sa1b1=(const float*)d_in[p++];
    const float *sa1W2=(const float*)d_in[p++], *sa1b2=(const float*)d_in[p++];
    const float *sa2W0=(const float*)d_in[p++], *sa2b0=(const float*)d_in[p++];
    const float *sa2W1=(const float*)d_in[p++], *sa2b1=(const float*)d_in[p++];
    const float *sa2W2=(const float*)d_in[p++], *sa2b2=(const float*)d_in[p++];
    const float *sa3W0=(const float*)d_in[p++], *sa3b0=(const float*)d_in[p++];
    const float *sa3W1=(const float*)d_in[p++], *sa3b1=(const float*)d_in[p++];
    const float *sa3W2=(const float*)d_in[p++], *sa3b2=(const float*)d_in[p++];
    const float *fp3W0=(const float*)d_in[p++], *fp3b0=(const float*)d_in[p++];
    const float *fp3W1=(const float*)d_in[p++], *fp3b1=(const float*)d_in[p++];
    const float *fp2W0=(const float*)d_in[p++], *fp2b0=(const float*)d_in[p++];
    const float *fp2W1=(const float*)d_in[p++], *fp2b1=(const float*)d_in[p++];
    const float *fp1W0=(const float*)d_in[p++], *fp1b0=(const float*)d_in[p++];
    const float *fp1W1=(const float*)d_in[p++], *fp1b1=(const float*)d_in[p++];
    const float *fp1W2=(const float*)d_in[p++], *fp1b2=(const float*)d_in[p++];
    const float *apW0 =(const float*)d_in[p++], *apb0 =(const float*)d_in[p++];
    const float *apW1 =(const float*)d_in[p++], *apb1 =(const float*)d_in[p++];
    const float *hdW0 =(const float*)d_in[p++], *hdb0 =(const float*)d_in[p++];
    const float *hdW1 =(const float*)d_in[p++], *hdb1 =(const float*)d_in[p++];
    const float *hdW2 =(const float*)d_in[p++], *hdb2 =(const float*)d_in[p++];
    const float *teW0 =(const float*)d_in[p++], *teb0 =(const float*)d_in[p++];
    const float *teW1 =(const float*)d_in[p++], *teb1 =(const float*)d_in[p++];
    const float *tpW0 =(const float*)d_in[p++], *tpb0 =(const float*)d_in[p++];
    const float *tpW1 =(const float*)d_in[p++], *tpb1 =(const float*)d_in[p++];
    const float *tpW2 =(const float*)d_in[p++], *tpb2 =(const float*)d_in[p++];

    const int P1 = BB * N1C;    // 6552
    const int P2 = BB * N2C;    // 1632
    const int PN = BB * NN;     // 32768

    float* ws = (float*)d_ws;
    size_t off = 0;
    auto alloc = [&](size_t n) { float* q = ws + off; off += n; return q; };

    float* pos1 = alloc((size_t)P1 * 3);
    int*   idx1 = (int*)alloc(P1);
    int*   nbr1 = (int*)alloc((size_t)P1 * KNBR);
    float* d21  = alloc((size_t)P1 * KNBR);
    float* x1   = alloc((size_t)P1 * 128);
    float* pos2 = alloc((size_t)P2 * 3);
    int*   idx2 = (int*)alloc(P2);
    int*   nbr2 = (int*)alloc((size_t)P2 * KNBR);
    float* d22  = alloc((size_t)P2 * KNBR);
    float* x2   = alloc((size_t)P2 * 256);
    float* cat3 = alloc((size_t)P2 * 259);
    float* t0   = alloc((size_t)P2 * 256);
    float* t1   = alloc((size_t)P2 * 512);
    float* t2   = alloc((size_t)P2 * 1024);
    float* g    = alloc((size_t)BB * 1024);
    float* fin3 = alloc((size_t)P2 * 1280);
    float* f3t  = alloc((size_t)P2 * 256);
    float* f3   = alloc((size_t)P2 * 256);
    float* i3   = alloc((size_t)P1 * 256);
    float* c2f  = alloc((size_t)P1 * 384);
    float* f2t  = alloc((size_t)P1 * 256);
    float* f2   = alloc((size_t)P1 * 128);
    float* i2   = alloc((size_t)PN * 128);
    float* c1f  = alloc((size_t)PN * 131);
    float* f1a  = alloc((size_t)PN * 128);
    float* f1b  = alloc((size_t)PN * 128);
    float* f1   = alloc((size_t)PN * 128);
    float* ap   = alloc((size_t)BB * 3);
    float* af0  = alloc((size_t)BB * 32);
    float* af   = alloc((size_t)BB * 64);
    float* hin  = alloc((size_t)PN * 192);
    float* h0   = alloc((size_t)PN * 128);
    float* h1   = alloc((size_t)PN * 128);
    float* hsc  = alloc((size_t)PN);
    float* gp   = alloc((size_t)BB * 6);
    float* cond = alloc((size_t)BB * 9);
    float* ae0  = alloc((size_t)BB * 32);
    float* af2  = alloc((size_t)BB * 64);
    float* tcat = alloc((size_t)BB * 1088);
    float* tp0  = alloc((size_t)BB * 512);
    float* tp1  = alloc((size_t)BB * 256);
    float* trn  = alloc((size_t)BB * 3);

    auto gemm = [&](const float* A, int M, int K, const float* Wm, const float* bb,
                    int Nc, float* Cm, int relu) {
        dim3 grd((M + 63) / 64, (Nc + 63) / 64);
        gemm_bias_act<<<grd, dim3(256), 0, stream>>>(A, Wm, bb, Cm, M, K, Nc, relu);
    };
    auto cat2 = [&](const float* A, int Ca, const float* Bp, int Cb, float* dst, int rows) {
        size_t tot = (size_t)rows * (Ca + Cb);
        concat2_kernel<<<dim3((tot + 255) / 256), dim3(256), 0, stream>>>(A, Ca, Bp, Cb, dst, rows);
    };

    const size_t saShmem = (size_t)2 * SA_ROWS * SA_STRIDE * sizeof(_Float16);

    // ---- SA1 ----
    fps_kernel<<<dim3(BB), dim3(256), 0, stream>>>(POS, NN, N1C, pos1, idx1);
    topk_kernel<<<dim3(BB * N1C), dim3(256), 0, stream>>>(POS, pos1, NN, N1C, d21, nbr1);
    sa_mlp_maxpool<<<dim3(BB * N1C), dim3(256), saShmem, stream>>>(
        POS, 3, POS, pos1, nbr1, d21, 0.2f * 0.2f, NN, N1C,
        sa1W0, sa1b0, 64, sa1W1, sa1b1, 64, sa1W2, sa1b2, 128, x1);

    // ---- SA2 ----
    fps_kernel<<<dim3(BB), dim3(256), 0, stream>>>(pos1, N1C, N2C, pos2, idx2);
    topk_kernel<<<dim3(BB * N2C), dim3(256), 0, stream>>>(pos1, pos2, N1C, N2C, d22, nbr2);
    sa_mlp_maxpool<<<dim3(BB * N2C), dim3(256), saShmem, stream>>>(
        x1, 128, pos1, pos2, nbr2, d22, 0.4f * 0.4f, N1C, N2C,
        sa2W0, sa2b0, 128, sa2W1, sa2b1, 128, sa2W2, sa2b2, 256, x2);

    // ---- SA3 / global feature ----
    cat2(x2, 256, pos2, 3, cat3, P2);
    gemm(cat3, P2, 259, sa3W0, sa3b0, 256, t0, 1);
    gemm(t0,   P2, 256, sa3W1, sa3b1, 512, t1, 1);
    gemm(t1,   P2, 512, sa3W2, sa3b2, 1024, t2, 0);
    rowmax_kernel<<<dim3(BB), dim3(256), 0, stream>>>(t2, N2C, 1024, g);

    // ---- FP3 ----
    {
        size_t tot = (size_t)P2 * 1280;
        concat_bcast_kernel<<<dim3((tot + 255) / 256), dim3(256), 0, stream>>>(
            x2, 256, g, 1024, N2C, 1, fin3, P2);
    }
    gemm(fin3, P2, 1280, fp3W0, fp3b0, 256, f3t, 1);
    gemm(f3t,  P2, 256,  fp3W1, fp3b1, 256, f3, 0);
    knn3_interp_kernel<<<dim3((BB * N1C + 255) / 256), dim3(256), 0, stream>>>(
        f3, pos2, pos1, N2C, N1C, 256, BB, i3);

    // ---- FP2 ----
    cat2(i3, 256, x1, 128, c2f, P1);
    gemm(c2f, P1, 384, fp2W0, fp2b0, 256, f2t, 1);
    gemm(f2t, P1, 256, fp2W1, fp2b1, 128, f2, 0);
    knn3_interp_kernel<<<dim3((BB * NN + 255) / 256), dim3(256), 0, stream>>>(
        f2, pos1, POS, N1C, NN, 128, BB, i2);

    // ---- FP1 ----
    cat2(i2, 128, POS, 3, c1f, PN);
    gemm(c1f, PN, 131, fp1W0, fp1b0, 128, f1a, 1);
    gemm(f1a, PN, 128, fp1W1, fp1b1, 128, f1b, 1);
    gemm(f1b, PN, 128, fp1W2, fp1b2, 128, f1, 0);

    // ---- approach-point encoder + head ----
    gather_ap_kernel<<<dim3(1), dim3(64), 0, stream>>>(POS, APIDX, NN, ap, BB);
    gemm(ap,  BB, 3,  apW0, apb0, 32, af0, 1);
    gemm(af0, BB, 32, apW1, apb1, 64, af, 0);
    {
        size_t tot = (size_t)PN * 192;
        concat_bcast_kernel<<<dim3((tot + 255) / 256), dim3(256), 0, stream>>>(
            f1, 128, af, 64, NN, 0, hin, PN);
    }
    gemm(hin, PN, 192, hdW0, hdb0, 128, h0, 1);
    gemm(h0,  PN, 128, hdW1, hdb1, 128, h1, 1);
    gemm(h1,  PN, 128, hdW2, hdb2, 1, hsc, 0);

    // ---- top-2 grasp points + conditioning ----
    top2_cond_kernel<<<dim3(BB), dim3(256), 0, stream>>>(hsc, POS, ap, NN, cond, gp);
    gemm(cond, BB, 9,  teW0, teb0, 32, ae0, 1);
    gemm(ae0,  BB, 32, teW1, teb1, 64, af2, 0);
    cat2(g, 1024, af2, 64, tcat, BB);
    gemm(tcat, BB, 1088, tpW0, tpb0, 512, tp0, 1);
    gemm(tp0,  BB, 512,  tpW1, tpb1, 256, tp1, 1);
    gemm(tp1,  BB, 256,  tpW2, tpb2, 3, trn, 0);

    grasp_finalize_kernel<<<dim3(1), dim3(64), 0, stream>>>(trn, gp, (float*)d_out, BB);
}
